// VanillaAttention_29600914604401
// MI455X (gfx1250) — matrix operand order repaired and checked by reading, not compiled, not run on hardware
//
#include <hip/hip_runtime.h>

typedef __attribute__((ext_vector_type(16))) _Float16     v16h;
typedef __attribute__((ext_vector_type(8)))  _Float16     v8h;
typedef __attribute__((ext_vector_type(8)))  float        v8f;
typedef __attribute__((ext_vector_type(4)))  float        f32x4;

#define B_   2
#define L_   2048
#define H_   16
#define E_   64
#define SROW (3 * H_ * E_)                 // float stride between seq positions
#define NKV  ((size_t)B_ * H_ * L_ * E_)   // elements in each f16 side buffer

static __device__ __forceinline__ v8f wmma16(v16h a, v16h b, v8f c) {
  // D = A(16x32 f16) * B(32x16 f16) + C(16x16 f32)
  return __builtin_amdgcn_wmma_f32_16x16x32_f16(
      /*neg_a=*/false, a, /*neg_b=*/false, b,
      /*c_mod=*/(short)0, c, /*reuse_a=*/false, /*reuse_b=*/false);
}

// ---------------------------------------------------------------------------
// Pass 1: K f32 -> f16, layout kf[b][h][s][e]  (B-fragment friendly, e-contig)
// ---------------------------------------------------------------------------
__global__ __launch_bounds__(256)
void cvt_k_kernel(const float* __restrict__ qkv, _Float16* __restrict__ kf) {
  const size_t t  = (size_t)blockIdx.x * 256 + threadIdx.x;  // 8 elems / thread
  const int    e  = (int)(t & 7) * 8;
  const size_t r  = t >> 3;                 // b*H*L + h*L + s
  const int    s  = (int)(r & (L_ - 1));
  const size_t bh = r >> 11;                // b*H + h
  const int    h  = (int)(bh & (H_ - 1));
  const int    b  = (int)(bh >> 4);

  const float* src = qkv + (((size_t)(b * L_ + s) * 3 + 1) * H_ + h) * E_ + e;
  f32x4 x = *(const f32x4*)(src);
  f32x4 y = *(const f32x4*)(src + 4);
  v8h o;
  o[0] = (_Float16)x[0]; o[1] = (_Float16)x[1]; o[2] = (_Float16)x[2]; o[3] = (_Float16)x[3];
  o[4] = (_Float16)y[0]; o[5] = (_Float16)y[1]; o[6] = (_Float16)y[2]; o[7] = (_Float16)y[3];
  *(v8h*)(kf + (bh * L_ + s) * E_ + e) = o;
}

// ---------------------------------------------------------------------------
// Pass 2: V f32 -> f16 transpose, layout vT[b][h][e][s]  (keys contiguous)
// ---------------------------------------------------------------------------
__global__ __launch_bounds__(256)
void vtrans_kernel(const float* __restrict__ qkv, _Float16* __restrict__ vT) {
  __shared__ __align__(16) _Float16 tile[64][40];   // [e][s], padded rows (80B)

  const int s0 = blockIdx.x * 32;
  const int h  = blockIdx.y;
  const int b  = blockIdx.z;
  const int tid = threadIdx.x;

  {
    const int srow = tid >> 3;            // 0..31
    const int e8   = (tid & 7) * 8;       // 0..56
    const float* src = qkv + (((size_t)(b * L_ + s0 + srow) * 3 + 2) * H_ + h) * E_ + e8;
    f32x4 x = *(const f32x4*)(src);
    f32x4 y = *(const f32x4*)(src + 4);
    tile[e8 + 0][srow] = (_Float16)x[0]; tile[e8 + 1][srow] = (_Float16)x[1];
    tile[e8 + 2][srow] = (_Float16)x[2]; tile[e8 + 3][srow] = (_Float16)x[3];
    tile[e8 + 4][srow] = (_Float16)y[0]; tile[e8 + 5][srow] = (_Float16)y[1];
    tile[e8 + 6][srow] = (_Float16)y[2]; tile[e8 + 7][srow] = (_Float16)y[3];
  }
  __syncthreads();
  {
    const int e    = tid >> 2;            // 0..63
    const int part = tid & 3;             // 0..3 -> 8 keys each
    v8h o = *(const v8h*)(&tile[e][part * 8]);
    const size_t bh = (size_t)b * H_ + h;
    *(v8h*)(vT + (bh * E_ + e) * L_ + s0 + part * 8) = o;
  }
}

// ---------------------------------------------------------------------------
// One 64-key flash-attention step (16 WMMA). MASKED only for diagonal steps.
// ---------------------------------------------------------------------------
template <bool MASKED>
static __device__ __forceinline__ void attn_step(
    int j, int q0, int h16, int n,
    const _Float16* __restrict__ kw, const _Float16* __restrict__ vw,
    _Float16 (*pt)[72], v16h aQ0, v16h aQ1,
    float (&rmax)[8], float (&rsum)[8], v8f (&acc)[4]) {

  // ---- S = Q K^T, four 16-key tiles (8 WMMA); K frags = direct 32B loads.
  float s[4][8];
#pragma unroll
  for (int t = 0; t < 4; ++t) {
    const _Float16* kr = kw + (size_t)(j + 16 * t + n) * E_ + 16 * h16;
    v16h b0 = *(const v16h*)(kr);        // e chunk 0..31
    v16h b1 = *(const v16h*)(kr + 32);   // e chunk 32..63
    v8f c;
#pragma unroll
    for (int r = 0; r < 8; ++r) c[r] = 0.0f;
    c = wmma16(aQ0, b0, c);
    c = wmma16(aQ1, b1, c);
#pragma unroll
    for (int r = 0; r < 8; ++r) s[t][r] = c[r];
  }

  // ---- Issue V fragment loads NOW; they complete under the softmax VALU.
  v16h vb[4][2];
#pragma unroll
  for (int c4 = 0; c4 < 4; ++c4) {
    const _Float16* vr = vw + (size_t)(16 * c4 + n) * L_ + j + 16 * h16;
    vb[c4][0] = *(const v16h*)(vr);        // keys j      + 16*h16 ..
    vb[c4][1] = *(const v16h*)(vr + 32);   // keys j + 32 + 16*h16 ..
  }

  // ---- (Masked-only) causal mask + online softmax over 64 columns.
  const int kg0 = j + n, kg1 = kg0 + 16, kg2 = kg0 + 32, kg3 = kg0 + 48;
#pragma unroll
  for (int r = 0; r < 8; ++r) {
    float v0 = s[0][r], v1 = s[1][r], v2 = s[2][r], v3 = s[3][r];
    if (MASKED) {
      const int mrow = q0 + 8 * h16 + r;
      v0 = (kg0 <= mrow) ? v0 : -1e30f;
      v1 = (kg1 <= mrow) ? v1 : -1e30f;
      v2 = (kg2 <= mrow) ? v2 : -1e30f;
      v3 = (kg3 <= mrow) ? v3 : -1e30f;
    }
    float mx = fmaxf(fmaxf(v0, v1), fmaxf(v2, v3));
    mx = fmaxf(mx, __shfl_xor(mx, 1));
    mx = fmaxf(mx, __shfl_xor(mx, 2));
    mx = fmaxf(mx, __shfl_xor(mx, 4));
    mx = fmaxf(mx, __shfl_xor(mx, 8));      // within the 16-lane half
    const float nm   = fmaxf(rmax[r], mx);
    const float corr = exp2f(rmax[r] - nm);
    rmax[r] = nm;
    const float p0 = exp2f(v0 - nm);
    const float p1 = exp2f(v1 - nm);
    const float p2 = exp2f(v2 - nm);
    const float p3 = exp2f(v3 - nm);
    rsum[r] = rsum[r] * corr + ((p0 + p1) + (p2 + p3));
    acc[0][r] *= corr; acc[1][r] *= corr; acc[2][r] *= corr; acc[3][r] *= corr;
    _Float16* pw = &pt[8 * h16 + r][n];
    pw[0]  = (_Float16)p0;  pw[16] = (_Float16)p1;
    pw[32] = (_Float16)p2;  pw[48] = (_Float16)p3;
  }

  // LDS in-order per wave; wait + clobber orders P store -> fragment read.
  asm volatile("s_wait_dscnt 0" ::: "memory");

  // ---- A = P fragments (16x64 -> two 16x32 A tiles), aligned 16B runs.
  v16h pa0, pa1;
  {
    const _Float16* pr = &pt[n][0];
    v8h a0 = *(const v8h*)(pr +      16 * h16);
    v8h a1 = *(const v8h*)(pr +  8 + 16 * h16);
    v8h a2 = *(const v8h*)(pr + 32 + 16 * h16);
    v8h a3 = *(const v8h*)(pr + 40 + 16 * h16);
#pragma unroll
    for (int i = 0; i < 8; ++i) {
      pa0[i] = a0[i]; pa0[8 + i] = a1[i];
      pa1[i] = a2[i]; pa1[8 + i] = a3[i];
    }
  }

  // ---- O += P V: 4 e-chunks x 2 key-chunks (8 WMMA), V already in regs.
#pragma unroll
  for (int c4 = 0; c4 < 4; ++c4) {
    acc[c4] = wmma16(pa0, vb[c4][0], acc[c4]);
    acc[c4] = wmma16(pa1, vb[c4][1], acc[c4]);
  }
  asm volatile("" ::: "memory");  // keep next iter's P stores behind these reads
}

// ---------------------------------------------------------------------------
// Main: per-wave flash attention, 64 keys / step, 16 WMMA / step.
// ---------------------------------------------------------------------------
__global__ __launch_bounds__(256)
void fa_fwd_kernel(const float* __restrict__ qkv,
                   const _Float16* __restrict__ kf,
                   const _Float16* __restrict__ vT,
                   float* __restrict__ out) {
  __shared__ __align__(16) _Float16 lds_p[8][16][72];  // per-wave P tile (16x64)

  const int wave = threadIdx.x >> 5;
  const int lane = threadIdx.x & 31;
  const int h16  = lane >> 4;
  const int n    = lane & 15;

  const int h  = blockIdx.y;
  const int b  = blockIdx.z;
  // Heaviest query blocks first: better GPU-wide load balance under causality.
  const int qb = (L_ / 128 - 1) - blockIdx.x;
  const int q0 = qb * 128 + wave * 16;

  _Float16 (*pt)[72] = lds_p[wave];
  const float lscale = 0.125f * 1.44269504088896340736f;  // 1/sqrt(E)*log2(e)

  const size_t bh = (size_t)b * H_ + h;
  const float*    qbase = qkv + ((size_t)b * L_ * 3) * (H_ * E_) + (size_t)h * E_;
  const _Float16* kw = kf + bh * L_ * E_;     // [s][e]
  const _Float16* vw = vT + bh * E_ * L_;     // [e][s]

  // ---- Q tile (16x64) -> two A fragments, scale folded into cvt.
  v16h aQ0, aQ1;
  {
    const float* qr = qbase + (size_t)(q0 + n) * SROW;
#define LD8(dst, off, ptr)                                                    \
    { f32x4 x = *(const f32x4*)(ptr); f32x4 y = *(const f32x4*)((ptr) + 4);   \
      dst[(off)+0] = (_Float16)(x[0]*lscale); dst[(off)+1] = (_Float16)(x[1]*lscale); \
      dst[(off)+2] = (_Float16)(x[2]*lscale); dst[(off)+3] = (_Float16)(x[3]*lscale); \
      dst[(off)+4] = (_Float16)(y[0]*lscale); dst[(off)+5] = (_Float16)(y[1]*lscale); \
      dst[(off)+6] = (_Float16)(y[2]*lscale); dst[(off)+7] = (_Float16)(y[3]*lscale); }
    LD8(aQ0, 0, qr +      16 * h16);
    LD8(aQ0, 8, qr +  8 + 16 * h16);
    LD8(aQ1, 0, qr + 32 + 16 * h16);
    LD8(aQ1, 8, qr + 40 + 16 * h16);
#undef LD8
  }

  float rmax[8], rsum[8];
  v8f acc[4];
#pragma unroll
  for (int r = 0; r < 8; ++r) {
    rmax[r] = -1e30f; rsum[r] = 0.0f;
    acc[0][r] = 0.0f; acc[1][r] = 0.0f; acc[2][r] = 0.0f; acc[3][r] = 0.0f;
  }

  const int nfull  = q0 >> 6;               // steps with no masking needed
  const int nsteps = (q0 + 16 + 63) >> 6;   // total 64-key steps (causal)

  for (int st = 0; st < nfull; ++st) {
    const int j = st * 64;
    __builtin_prefetch(kw + (size_t)(j + 64 + n) * E_, 0, 0);
    __builtin_prefetch(vw + (size_t)n * L_ + j + 64, 0, 0);
    attn_step<false>(j, q0, h16, n, kw, vw, pt, aQ0, aQ1, rmax, rsum, acc);
  }
  for (int st = nfull; st < nsteps; ++st) {
    attn_step<true>(st * 64, q0, h16, n, kw, vw, pt, aQ0, aQ1, rmax, rsum, acc);
  }

  // ---- Finalize: reduce row sums, normalize, store.
#pragma unroll
  for (int r = 0; r < 8; ++r) {
    float sg = rsum[r];
    sg += __shfl_xor(sg, 1);
    sg += __shfl_xor(sg, 2);
    sg += __shfl_xor(sg, 4);
    sg += __shfl_xor(sg, 8);
    rsum[r] = 1.0f / sg;
  }
  float* ob = out + (((size_t)b * L_ + q0) * H_ + h) * E_;
#pragma unroll
  for (int c4 = 0; c4 < 4; ++c4) {
#pragma unroll
    for (int r = 0; r < 8; ++r) {
      const int M = 8 * h16 + r;
      ob[(size_t)M * (H_ * E_) + 16 * c4 + n] = acc[c4][r] * rsum[r];
    }
  }
}

extern "C" void kernel_launch(void* const* d_in, const int* in_sizes, int n_in,
                              void* d_out, int out_size, void* d_ws, size_t ws_size,
                              hipStream_t stream) {
  (void)in_sizes; (void)n_in; (void)out_size; (void)ws_size;
  const float* qkv = (const float*)d_in[0];
  float* out = (float*)d_out;

  _Float16* kf = (_Float16*)d_ws;          // NKV halves
  _Float16* vT = kf + NKV;                 // NKV halves  (total 16.8 MB of ws)

  cvt_k_kernel<<<dim3((unsigned)(NKV / 8 / 256)), dim3(256), 0, stream>>>(qkv, kf);
  vtrans_kernel<<<dim3(L_ / 32, H_, B_), dim3(256), 0, stream>>>(qkv, vT);
  fa_fwd_kernel<<<dim3(L_ / 128, H_, B_), dim3(256), 0, stream>>>(qkv, kf, vT, out);
}
